// MAB_48507360641094
// MI455X (gfx1250) — compile-verified, matrix operand order checked by reading, not run on hardware
//
#include <hip/hip_runtime.h>
#include <hip/hip_bf16.h>

// ---------------------------------------------------------------------------
// DANet-style dual attention (CAM + SAM) fused for MI455X (gfx1250, wave32).
// All GEMMs use v_wmma_f32_16x16x32_bf16. SAM uses flash-attention online
// softmax so the 4096x4096 energy matrix (537MB fp32 over all 8 batches) is never materialized.
// Scratch (d_ws) requirement: 24 MiB.
// ---------------------------------------------------------------------------

typedef __attribute__((ext_vector_type(16))) __bf16 v16bf;
typedef __attribute__((ext_vector_type(8)))  float  v8f;

#define CC   64
#define HW   4096
#define BB   8

static __device__ __forceinline__ v8f zero8() {
    v8f z;
#pragma unroll
    for (int i = 0; i < 8; ++i) z[i] = 0.0f;
    return z;
}

static __device__ __forceinline__ v8f wmma_bf16(v16bf a, v16bf b, v8f c) {
    // (neg_a, A, neg_b, B, c_mod, C, reuse_a, reuse_b)
    return __builtin_amdgcn_wmma_f32_16x16x32_bf16(false, a, false, b,
                                                   (short)0, c, false, false);
}

// K index for element pair j (0..7) of an A/B fragment, given lane-half hi.
// Per ISA 16-bit A layout: vgpr j<4 -> K = 2j + 8*hi, j>=4 -> K = 16 + 2(j-4) + 8*hi.
static __device__ __forceinline__ int kofs(int j, int hi) {
    return ((j < 4) ? (2 * j) : (16 + 2 * (j - 4))) + 8 * hi;
}

// Load a 16x32 A (or B, symmetric layout) fragment from memory where this
// lane's row (A: m = lane&15; B: n = lane&15) is contiguous in K.
// p must already point at &M[row][k_base].
static __device__ __forceinline__ v16bf load_frag(const __bf16* p, int hi) {
    v16bf f;
#pragma unroll
    for (int j = 0; j < 8; ++j) {
        int k0 = kofs(j, hi);
        f[2 * j]     = p[k0];
        f[2 * j + 1] = p[k0 + 1];
    }
    return f;
}

// Same, converting fp32 memory -> bf16 fragment on the fly (for weights).
static __device__ __forceinline__ v16bf load_frag_f32(const float* p, int hi) {
    v16bf f;
#pragma unroll
    for (int j = 0; j < 8; ++j) {
        int k0 = kofs(j, hi);
        f[2 * j]     = (__bf16)p[k0];
        f[2 * j + 1] = (__bf16)p[k0 + 1];
    }
    return f;
}

// ---------------------------------------------------------------------------
// K0: convert + transpose x[b][c][n] (f32) -> x_t[b][n][c] (bf16)
// ---------------------------------------------------------------------------
__global__ __launch_bounds__(256) void k_xt(const float* __restrict__ x,
                                            __bf16* __restrict__ x_t) {
    int idx = blockIdx.x * 256 + threadIdx.x;      // exactly B*C*HW threads
    int n = idx & (HW - 1);
    int c = (idx >> 12) & (CC - 1);
    int b = idx >> 18;
    x_t[((size_t)b * HW + n) * CC + c] = (__bf16)x[idx];
}

// ---------------------------------------------------------------------------
// K1: all four 1x1-conv projections as WMMA GEMMs (M x 4096, K = 64).
//   xc, v  stored [c][hw]  (K-contiguous for energy / PV B-fragments)
//   q, k   stored [n][32]  (K-contiguous over channels for S fragments)
// ---------------------------------------------------------------------------
__global__ __launch_bounds__(128) void k_proj(const float* __restrict__ w_cam,
                                              const float* __restrict__ w_q,
                                              const float* __restrict__ w_k,
                                              const float* __restrict__ w_v,
                                              const __bf16* __restrict__ x_t,
                                              __bf16* __restrict__ xc,
                                              __bf16* __restrict__ q_t,
                                              __bf16* __restrict__ k_t,
                                              __bf16* __restrict__ vbf) {
    int flat = blockIdx.x * 4 + (threadIdx.x >> 5);   // 24576 waves total
    int b    = flat / 3072;
    int rem  = flat % 3072;
    int nt   = rem / 12;
    int slot = rem % 12;
    int lane = threadIdx.x & 31, ln = lane & 15, hi = lane >> 4;

    const float* W; __bf16* outp; int m0; int chlay;
    if (slot < 4)       { W = w_cam; outp = xc;  m0 = slot * 16;        chlay = 1; }
    else if (slot < 8)  { W = w_v;   outp = vbf; m0 = (slot - 4) * 16;  chlay = 1; }
    else if (slot < 10) { W = w_q;   outp = q_t; m0 = (slot - 8) * 16;  chlay = 0; }
    else                { W = w_k;   outp = k_t; m0 = (slot - 10) * 16; chlay = 0; }

    int n = nt * 16 + ln;
    const __bf16* xb = x_t + ((size_t)b * HW + n) * CC;
    const float*  wr = W + (m0 + ln) * CC;

    v8f acc = zero8();
#pragma unroll
    for (int kc = 0; kc < CC; kc += 32) {
        v16bf a  = load_frag_f32(wr + kc, hi);
        v16bf bm = load_frag(xb + kc, hi);
        acc = wmma_bf16(a, bm, acc);
    }

    if (chlay) {
        __bf16* o = outp + (size_t)b * CC * HW;
#pragma unroll
        for (int j = 0; j < 8; ++j)
            o[(size_t)(m0 + j + 8 * hi) * HW + n] = (__bf16)acc[j];
    } else {
        __bf16* o = outp + (size_t)b * HW * 32;
#pragma unroll
        for (int j = 0; j < 8; ++j)
            o[(size_t)n * 32 + m0 + j + 8 * hi] = (__bf16)acc[j];
    }
}

// ---------------------------------------------------------------------------
// K2: CAM. One 256-thread block per batch.
//   energy = xc.xc^T (64x64, K=4096, WMMA), softmax rows in LDS,
//   out_c = attn_c . x  -> cat_t[b][n][0:64] (bf16)
// ---------------------------------------------------------------------------
__global__ __launch_bounds__(256) void k_cam(const __bf16* __restrict__ xc,
                                             const __bf16* __restrict__ x_t,
                                             __bf16* __restrict__ cat_t) {
    __shared__ float  e[64][64];
    __shared__ __bf16 attn[64][64];
    int b = blockIdx.x;
    int tid = threadIdx.x, wv = tid >> 5, lane = tid & 31, ln = lane & 15, hi = lane >> 4;
    const __bf16* xcb = xc + (size_t)b * CC * HW;

    // energy: 16 tiles of 16x16 over 8 waves
#pragma unroll
    for (int t = wv * 2; t < wv * 2 + 2; ++t) {
        int mi = t >> 2, ni = t & 3;
        const __bf16* ra = xcb + (size_t)(mi * 16 + ln) * HW;
        const __bf16* rb = xcb + (size_t)(ni * 16 + ln) * HW;
        v8f acc = zero8();
        for (int kc = 0; kc < HW; kc += 32) {
            v16bf a  = load_frag(ra + kc, hi);
            v16bf bm = load_frag(rb + kc, hi);
            acc = wmma_bf16(a, bm, acc);
        }
#pragma unroll
        for (int j = 0; j < 8; ++j)
            e[mi * 16 + j + 8 * hi][ni * 16 + ln] = acc[j];
    }
    __syncthreads();

    if (tid < 64) {
        float mx = -3.0e38f;
        for (int d = 0; d < 64; ++d) mx = fmaxf(mx, e[tid][d]);
        float s = 0.0f;
        for (int d = 0; d < 64; ++d) { float p = __expf(e[tid][d] - mx); e[tid][d] = p; s += p; }
        float inv = 1.0f / s;
        for (int d = 0; d < 64; ++d) attn[tid][d] = (__bf16)(e[tid][d] * inv);
    }
    __syncthreads();

    // out_c = attn (64x64) . x (64x4096): 1024 tiles over 8 waves
    __bf16* o = cat_t + (size_t)b * HW * 128;
    for (int t = wv; t < 1024; t += 8) {
        int mi = t & 3, nt = t >> 2;
        int n = nt * 16 + ln;
        const __bf16* xb = x_t + ((size_t)b * HW + n) * CC;
        v8f acc = zero8();
#pragma unroll
        for (int kc = 0; kc < CC; kc += 32) {
            v16bf a  = load_frag(&attn[mi * 16 + ln][kc], hi);
            v16bf bm = load_frag(xb + kc, hi);
            acc = wmma_bf16(a, bm, acc);
        }
#pragma unroll
        for (int j = 0; j < 8; ++j)
            o[(size_t)n * 128 + mi * 16 + j + 8 * hi] = (__bf16)acc[j];
    }
}

// ---------------------------------------------------------------------------
// K3: SAM flash attention. Each wave owns 32 query rows (2 row-tiles),
// sweeps 4096 keys in 64-wide chunks. S: 4 WMMAs per row-tile (K=32 = one
// instr each). Online softmax per row (rows of a C-fragment live in one
// 16-lane half -> shfl_xor(1,2,4,8) reductions). Both row-tiles' P matrices
// are staged in a 32x64 per-wave LDS tile (C->A layout relayout), then one
// PV pass loads each V fragment ONCE and feeds both row-tiles' accumulators
// (16 WMMAs per chunk, V gather traffic halved vs per-row reload).
// Output -> cat_t[b][n][64:128].
// ---------------------------------------------------------------------------
__global__ __launch_bounds__(128) void k_sam(const __bf16* __restrict__ q_t,
                                             const __bf16* __restrict__ k_t,
                                             const __bf16* __restrict__ vbf,
                                             __bf16* __restrict__ cat_t) {
    __shared__ __bf16 pbuf[4][32 * 64];
    int flat = blockIdx.x * 4 + (threadIdx.x >> 5);    // 1024 waves
    int b  = flat >> 7;
    int rb = flat & 127;
    int n0 = rb * 32;
    int lane = threadIdx.x & 31, ln = lane & 15, hi = lane >> 4;
    __bf16* myp = pbuf[threadIdx.x >> 5];

    const __bf16* qb = q_t + (size_t)b * HW * 32;
    const __bf16* kb = k_t + (size_t)b * HW * 32;
    const __bf16* vb = vbf + (size_t)b * CC * HW;

    v16bf Aq[2];
#pragma unroll
    for (int r = 0; r < 2; ++r)
        Aq[r] = load_frag(qb + (size_t)(n0 + 16 * r + ln) * 32, hi);

    float mrow[2][8], lrow[2][8];
    v8f   O[2][4];
#pragma unroll
    for (int r = 0; r < 2; ++r)
#pragma unroll
        for (int j = 0; j < 8; ++j) { mrow[r][j] = -3.0e38f; lrow[r][j] = 0.0f; }
#pragma unroll
    for (int r = 0; r < 2; ++r)
#pragma unroll
        for (int ct = 0; ct < 4; ++ct) O[r][ct] = zero8();

    for (int mc = 0; mc < HW / 64; ++mc) {
        int m0 = mc * 64;
        v16bf B0 = load_frag(kb + (size_t)(m0 + ln) * 32, hi);
        v16bf B1 = load_frag(kb + (size_t)(m0 + 16 + ln) * 32, hi);
        v16bf B2 = load_frag(kb + (size_t)(m0 + 32 + ln) * 32, hi);
        v16bf B3 = load_frag(kb + (size_t)(m0 + 48 + ln) * 32, hi);

        // ---- softmax for both row-tiles, stage P in LDS ----
#pragma unroll
        for (int r = 0; r < 2; ++r) {
            v8f s0 = wmma_bf16(Aq[r], B0, zero8());
            v8f s1 = wmma_bf16(Aq[r], B1, zero8());
            v8f s2 = wmma_bf16(Aq[r], B2, zero8());
            v8f s3 = wmma_bf16(Aq[r], B3, zero8());

            float nm[8], sc[8];
#pragma unroll
            for (int j = 0; j < 8; ++j) {
                float t = fmaxf(fmaxf(s0[j], s1[j]), fmaxf(s2[j], s3[j]));
                t = fmaxf(t, __shfl_xor(t, 1));
                t = fmaxf(t, __shfl_xor(t, 2));
                t = fmaxf(t, __shfl_xor(t, 4));
                t = fmaxf(t, __shfl_xor(t, 8));
                nm[j] = fmaxf(mrow[r][j], t);
            }
#pragma unroll
            for (int j = 0; j < 8; ++j) {
                float p0 = __expf(s0[j] - nm[j]);
                float p1 = __expf(s1[j] - nm[j]);
                float p2 = __expf(s2[j] - nm[j]);
                float p3 = __expf(s3[j] - nm[j]);
                float rs = p0 + p1 + p2 + p3;
                rs += __shfl_xor(rs, 1);
                rs += __shfl_xor(rs, 2);
                rs += __shfl_xor(rs, 4);
                rs += __shfl_xor(rs, 8);
                float scale = __expf(mrow[r][j] - nm[j]);
                lrow[r][j] = lrow[r][j] * scale + rs;
                mrow[r][j] = nm[j];
                sc[j] = scale;
                int row = r * 16 + j + 8 * hi;
                myp[row * 64 + ln]      = (__bf16)p0;
                myp[row * 64 + 16 + ln] = (__bf16)p1;
                myp[row * 64 + 32 + ln] = (__bf16)p2;
                myp[row * 64 + 48 + ln] = (__bf16)p3;
            }
#pragma unroll
            for (int ct = 0; ct < 4; ++ct)
#pragma unroll
                for (int j = 0; j < 8; ++j) O[r][ct][j] *= sc[j];
        }

        // LDS pipe is in-order per wave; fence the compiler + wait dscnt.
        asm volatile("s_wait_dscnt 0x0" ::: "memory");
        v16bf P[2][2];
#pragma unroll
        for (int r = 0; r < 2; ++r) {
            P[r][0] = load_frag(myp + (r * 16 + ln) * 64, hi);
            P[r][1] = load_frag(myp + (r * 16 + ln) * 64 + 32, hi);
        }
        asm volatile("" ::: "memory");

        // ---- PV: each V fragment loaded once, used by both row-tiles ----
#pragma unroll
        for (int ct = 0; ct < 4; ++ct) {
            const __bf16* vr = vb + (size_t)(ct * 16 + ln) * HW + m0;
            v16bf V0 = load_frag(vr, hi);
            v16bf V1 = load_frag(vr + 32, hi);
            O[0][ct] = wmma_bf16(P[0][0], V0, O[0][ct]);
            O[0][ct] = wmma_bf16(P[0][1], V1, O[0][ct]);
            O[1][ct] = wmma_bf16(P[1][0], V0, O[1][ct]);
            O[1][ct] = wmma_bf16(P[1][1], V1, O[1][ct]);
        }
    }

    __bf16* o = cat_t + (size_t)b * HW * 128;
#pragma unroll
    for (int r = 0; r < 2; ++r)
#pragma unroll
        for (int ct = 0; ct < 4; ++ct)
#pragma unroll
            for (int j = 0; j < 8; ++j) {
                float val = O[r][ct][j] / lrow[r][j];
                int n = n0 + 16 * r + j + 8 * hi;
                o[(size_t)n * 128 + 64 + ct * 16 + ln] = (__bf16)val;
            }
}

// ---------------------------------------------------------------------------
// K4: bottleneck 1x1 conv (K = 128) + residual, fp32 output.
// ---------------------------------------------------------------------------
__global__ __launch_bounds__(128) void k_bn(const float* __restrict__ w_bn,
                                            const __bf16* __restrict__ cat_t,
                                            const float* __restrict__ x,
                                            float* __restrict__ out) {
    int flat = blockIdx.x * 4 + (threadIdx.x >> 5);    // 8192 waves
    int b   = flat >> 10;
    int rem = flat & 1023;
    int mi  = rem & 3;
    int nt  = rem >> 2;
    int lane = threadIdx.x & 31, ln = lane & 15, hi = lane >> 4;
    int m0 = mi * 16, n = nt * 16 + ln;

    const float*  wr = w_bn + (m0 + ln) * 128;
    const __bf16* cb = cat_t + ((size_t)b * HW + n) * 128;

    v8f acc = zero8();
#pragma unroll
    for (int kc = 0; kc < 128; kc += 32) {
        v16bf a  = load_frag_f32(wr + kc, hi);
        v16bf bm = load_frag(cb + kc, hi);
        acc = wmma_bf16(a, bm, acc);
    }
#pragma unroll
    for (int j = 0; j < 8; ++j) {
        size_t idx = ((size_t)b * CC + m0 + j + 8 * hi) * HW + n;
        out[idx] = x[idx] + acc[j];
    }
}

// ---------------------------------------------------------------------------
extern "C" void kernel_launch(void* const* d_in, const int* in_sizes, int n_in,
                              void* d_out, int out_size, void* d_ws, size_t ws_size,
                              hipStream_t stream) {
    const float* x     = (const float*)d_in[0];
    const float* w_cam = (const float*)d_in[1];
    const float* w_q   = (const float*)d_in[2];
    const float* w_k   = (const float*)d_in[3];
    const float* w_v   = (const float*)d_in[4];
    const float* w_bn  = (const float*)d_in[5];
    float* out = (float*)d_out;

    char* ws = (char*)d_ws;                       // 24 MiB used
    __bf16* x_t   = (__bf16*)(ws);                // [8][4096][64]  4 MiB
    __bf16* xc    = (__bf16*)(ws + (4u  << 20));  // [8][64][4096]  4 MiB
    __bf16* q_t   = (__bf16*)(ws + (8u  << 20));  // [8][4096][32]  2 MiB
    __bf16* k_t   = (__bf16*)(ws + (10u << 20));  // [8][4096][32]  2 MiB
    __bf16* vbf   = (__bf16*)(ws + (12u << 20));  // [8][64][4096]  4 MiB
    __bf16* cat_t = (__bf16*)(ws + (16u << 20));  // [8][4096][128] 8 MiB

    k_xt  <<<(BB * CC * HW) / 256, 256, 0, stream>>>(x, x_t);
    k_proj<<<6144, 128, 0, stream>>>(w_cam, w_q, w_k, w_v, x_t, xc, q_t, k_t, vbf);
    k_cam <<<BB, 256, 0, stream>>>(xc, x_t, cat_t);
    k_sam <<<256, 128, 0, stream>>>(q_t, k_t, vbf, cat_t);
    k_bn  <<<2048, 128, 0, stream>>>(w_bn, cat_t, x, out);
}
